// LearnedSimulator_66855460930204
// MI455X (gfx1250) — matrix operand order repaired and checked by reading, not compiled, not run on hardware
//
#include <hip/hip_runtime.h>
#include <hip/hip_bf16.h>

// ---------------- problem constants ----------------
#define JOINTS 22
#define BATCH  256
#define NNODE  (BATCH * JOINTS)            // 5632
#define NEDGE  (BATCH * JOINTS * JOINTS)   // 123904  (= 484 per graph)
#define LAT    128
#define NSTEPS 10
#define LN_EPS 1e-5f

typedef __attribute__((ext_vector_type(8)))  __bf16 bf16x8_t;
typedef __attribute__((ext_vector_type(16))) __bf16 bf16x16_t;
typedef __attribute__((ext_vector_type(8)))  float  f32x8_t;
typedef __attribute__((ext_vector_type(4)))  float  f32x4_t;

// Convert 8 consecutive f32 -> packed bf16x8 (single 16B LDS store at the call site)
__device__ __forceinline__ bf16x8_t cvt8(const float* src) {
    f32x4_t f0 = *(const f32x4_t*)(src);
    f32x4_t f1 = *(const f32x4_t*)(src + 4);
    bf16x8_t d;
#pragma unroll
    for (int u = 0; u < 4; ++u) { d[u] = (__bf16)f0[u]; d[4 + u] = (__bf16)f1[u]; }
    return d;
}

// ---------------- WMMA fragment loaders (CDNA5 16x16x32 bf16, wave32) ------
// A 16x32 (MxK): lane<16 -> M=lane, K in {k0..k0+7} U {k0+16..k0+23}
//                lane>=16 -> M=lane-16, K in {k0+8..k0+15} U {k0+24..k0+31}
__device__ __forceinline__ bf16x16_t load_a_frag(const __bf16* rowp, int k0, int lane) {
    const __bf16* p = rowp + k0 + ((lane & 16) ? 8 : 0);
    bf16x8_t lo = *(const bf16x8_t*)(p);
    bf16x8_t hi = *(const bf16x8_t*)(p + 16);
    bf16x16_t r;
#pragma unroll
    for (int u = 0; u < 8; ++u) { r[u] = lo[u]; r[u + 8] = hi[u]; }
    return r;
}
// B 32x16 (KxN), stored as W^T (N rows x K cols, row-major, contiguous K):
// lane<16 -> N=lane, K in {k0..k0+7} U {k0+16..k0+23} ; lane>=16 -> N=lane-16, K in {k0+8..k0+15} U {k0+24..k0+31}
__device__ __forceinline__ bf16x16_t load_b_frag(const __bf16* colp, int k0, int lane) {
    const __bf16* p = colp + k0 + ((lane & 16) ? 8 : 0);
    bf16x8_t lo = *(const bf16x8_t*)(p);
    bf16x8_t hi = *(const bf16x8_t*)(p + 16);
    bf16x16_t r;
#pragma unroll
    for (int u = 0; u < 8; ++u) { r[u] = lo[u]; r[u + 8] = hi[u]; }
    return r;
}

// One wave computes a 16x32 output tile: rows [rowBase,rowBase+16), cols [colBase,colBase+32)
template <int K>
__device__ __forceinline__ void gemm_tile(const __bf16* Xbase, int xStride, const __bf16* wT,
                                          int lane, int rowBase, int colBase,
                                          f32x8_t& a0, f32x8_t& a1) {
    static_assert(K % 32 == 0, "K must be multiple of 32");
    const __bf16* rowp = Xbase + (size_t)(rowBase + (lane & 15)) * xStride;
    const __bf16* c0   = wT + (size_t)(colBase + (lane & 15)) * K;
    const __bf16* c1   = c0 + (size_t)16 * K;
#pragma unroll
    for (int k0 = 0; k0 < K; k0 += 32) {
        bf16x16_t a  = load_a_frag(rowp, k0, lane);
        bf16x16_t b0 = load_b_frag(c0, k0, lane);
        a0 = __builtin_amdgcn_wmma_f32_16x16x32_bf16(false, a, false, b0, (short)0, a0, false, false);
        bf16x16_t b1 = load_b_frag(c1, k0, lane);
        a1 = __builtin_amdgcn_wmma_f32_16x16x32_bf16(false, a, false, b1, (short)0, a1, false, false);
    }
}

// C/D layout: VGPR v, lane -> M = v + 8*(lane>>4), N = lane&15
__device__ __forceinline__ void store_relu_h(__bf16* H, const f32x8_t& a0, const f32x8_t& a1,
                                             const float* bias, int lane, int rowBase, int colBase) {
    int rr = rowBase + ((lane & 16) ? 8 : 0);
    int cc = colBase + (lane & 15);
#pragma unroll
    for (int nt = 0; nt < 2; ++nt) {
        int col = cc + nt * 16;
        float bv = bias[col];
        const f32x8_t& a = nt ? a1 : a0;
#pragma unroll
        for (int v = 0; v < 8; ++v) {
            float x = a[v] + bv;
            H[(rr + v) * LAT + col] = (__bf16)(x > 0.f ? x : 0.f);
        }
    }
}

enum { MODE_GENERIC = 0, MODE_EDGE = 1, MODE_NODE = 2 };

// Fused 3-layer MLP (K1 -> 128 -> 128 -> 128) + LayerNorm epilogue (+mask, +residual).
// Block: 256 threads (8 waves) computing 32 rows x 128 cols. Hidden stays in LDS.
template <int MODE, int K1>
__global__ __launch_bounds__(256) void mlp3_kernel(
    const float* X0, const float* X1, int Kreal,
    const __bf16* w1, const float* b1,
    const __bf16* w2, const float* b2,
    const __bf16* w3, const float* b3,
    const float* lng, const float* lnb,
    const float* maskp, const float* resp,
    float* out) {
    extern __shared__ char smem[];
    constexpr int R0 = ((32 * K1 * 2) > 16384 ? (32 * K1 * 2) : 16384);
    __bf16* Xs  = (__bf16*)smem;          // 32 x K1 bf16
    float*  O   = (float*)smem;            // 32 x 128 f32 (aliases Xs; used after GEMM1)
    __bf16* H   = (__bf16*)(smem + R0);    // 32 x 128 bf16
    float* redS  = (float*)(smem + R0 + 32 * LAT * 2);
    float* redS2 = redS + 32 * 8;
    float* mr    = redS2 + 32 * 8;

    const int tid = threadIdx.x;
    const int lane = tid & 31;
    const int waveId = tid >> 5;
    const int rowBase = (waveId >> 2) * 16;   // 0 or 16
    const int colBase = (waveId & 3) * 32;    // 0,32,64,96
    const int blockRow = blockIdx.x * 32;

    // ---- stage X tile into LDS as bf16 (gather / concat / zero-pad), 16B stores ----
    if (MODE == MODE_EDGE) {
        const int CH = K1 / 8;   // 48 chunks of 8
        for (int idx = tid; idx < 32 * CH; idx += 256) {
            int r  = idx / CH;
            int c8 = (idx - r * CH) * 8;
            int e  = blockRow + r;
            int b  = e / 484; int rem = e - b * 484;
            int ii = rem / 22; int jj = rem - ii * 22;
            const float* src; int c;
            if (c8 < 128)      { src = X0 + (size_t)e * LAT;               c = c8; }
            else if (c8 < 256) { src = X1 + (size_t)(b * 22 + ii) * LAT;   c = c8 - 128; }
            else               { src = X1 + (size_t)(b * 22 + jj) * LAT;   c = c8 - 256; }
            *(bf16x8_t*)(Xs + r * K1 + c8) = cvt8(src + c);
        }
    } else if (MODE == MODE_NODE) {
        const int CH = K1 / 8;   // 32
        for (int idx = tid; idx < 32 * CH; idx += 256) {
            int r  = idx / CH;
            int c8 = (idx - r * CH) * 8;
            int n  = blockRow + r;
            const float* src; int c;
            if (c8 < 128) { src = X0 + (size_t)n * LAT; c = c8; }
            else          { src = X1 + (size_t)n * LAT; c = c8 - 128; }
            *(bf16x8_t*)(Xs + r * K1 + c8) = cvt8(src + c);
        }
    } else {
        for (int idx = tid; idx < 32 * K1; idx += 256) {
            int r = idx / K1; int k = idx - r * K1;
            int grow = blockRow + r;
            float v = (k < Kreal) ? X0[(size_t)grow * Kreal + k] : 0.f;
            Xs[idx] = (__bf16)v;
        }
    }
    __syncthreads();

    const f32x8_t Z = {0.f, 0.f, 0.f, 0.f, 0.f, 0.f, 0.f, 0.f};
    f32x8_t a0 = Z, a1 = Z;

    // layer 1
    gemm_tile<K1>(Xs, K1, w1, lane, rowBase, colBase, a0, a1);
    store_relu_h(H, a0, a1, b1, lane, rowBase, colBase);
    __syncthreads();

    // layer 2
    a0 = Z; a1 = Z;
    gemm_tile<LAT>(H, LAT, w2, lane, rowBase, colBase, a0, a1);
    __syncthreads();                 // all reads of H done before overwrite
    store_relu_h(H, a0, a1, b2, lane, rowBase, colBase);
    __syncthreads();

    // layer 3 -> O (f32)
    a0 = Z; a1 = Z;
    gemm_tile<LAT>(H, LAT, w3, lane, rowBase, colBase, a0, a1);
    {
        int rr = rowBase + ((lane & 16) ? 8 : 0);
        int cc = colBase + (lane & 15);
#pragma unroll
        for (int nt = 0; nt < 2; ++nt) {
            int col = cc + nt * 16;
            float bv = b3[col];
            const f32x8_t& a = nt ? a1 : a0;
#pragma unroll
            for (int v = 0; v < 8; ++v) O[(rr + v) * LAT + col] = a[v] + bv;
        }
    }
    __syncthreads();

    // ---- fused LayerNorm (+mask, +residual) epilogue ----
    {
        int row = tid >> 3;
        int part = tid & 7;
        const float* orow = O + row * LAT;
        float s = 0.f, s2 = 0.f;
#pragma unroll
        for (int c = part * 16; c < part * 16 + 16; ++c) { float v = orow[c]; s += v; s2 += v * v; }
        redS[row * 8 + part] = s; redS2[row * 8 + part] = s2;
        __syncthreads();
        if (part == 0) {
            float S = 0.f, S2 = 0.f;
#pragma unroll
            for (int p = 0; p < 8; ++p) { S += redS[row * 8 + p]; S2 += redS2[row * 8 + p]; }
            float m = S * (1.f / 128.f);
            float var = S2 * (1.f / 128.f) - m * m;
            mr[row * 2] = m; mr[row * 2 + 1] = rsqrtf(var + LN_EPS);
        }
        __syncthreads();
        float m = mr[row * 2], rs = mr[row * 2 + 1];
        int grow = blockRow + row;
        float mk = maskp ? maskp[grow] : 1.f;
        size_t obase = (size_t)grow * LAT;
#pragma unroll
        for (int c = part * 16; c < part * 16 + 16; ++c) {
            float y = (orow[c] - m) * rs * lng[c] + lnb[c];
            y *= mk;
            if (resp) y += resp[obase + c];
            out[obase + c] = y;
        }
    }
}

// Decoder: 128 ->(WMMA) 128 ->(WMMA) 128 ->(scalar) 3, then out = pos_last + pred
__global__ __launch_bounds__(256) void decoder_kernel(
    const float* node,
    const __bf16* w1, const float* b1,
    const __bf16* w2, const float* b2,
    const float* W3, const float* b3,   // f32 (128x3), bias 3
    const float* pos, float* out) {
    __shared__ __bf16 Xs[32 * LAT];
    __shared__ __bf16 H[32 * LAT];
    const int tid = threadIdx.x;
    const int lane = tid & 31;
    const int waveId = tid >> 5;
    const int rowBase = (waveId >> 2) * 16;
    const int colBase = (waveId & 3) * 32;
    const int blockRow = blockIdx.x * 32;

    for (int idx = tid; idx < 32 * (LAT / 8); idx += 256) {
        int r = idx >> 4; int c8 = (idx & 15) * 8;
        *(bf16x8_t*)(Xs + r * LAT + c8) = cvt8(node + (size_t)(blockRow + r) * LAT + c8);
    }
    __syncthreads();

    const f32x8_t Z = {0.f, 0.f, 0.f, 0.f, 0.f, 0.f, 0.f, 0.f};
    f32x8_t a0 = Z, a1 = Z;
    gemm_tile<LAT>(Xs, LAT, w1, lane, rowBase, colBase, a0, a1);
    store_relu_h(H, a0, a1, b1, lane, rowBase, colBase);
    __syncthreads();
    a0 = Z; a1 = Z;
    gemm_tile<LAT>(H, LAT, w2, lane, rowBase, colBase, a0, a1);
    __syncthreads();
    store_relu_h(Xs, a0, a1, b2, lane, rowBase, colBase);   // h2 -> Xs (Xs dead)
    __syncthreads();

    for (int idx = tid; idx < 32 * 3; idx += 256) {
        int r = idx / 3; int d = idx - r * 3;
        float s = b3[d];
        for (int c = 0; c < LAT; ++c) s += (float)Xs[r * LAT + c] * W3[c * 3 + d];
        int grow = blockRow + r;
        out[(size_t)grow * 3 + d] = pos[(size_t)grow * 18 + 15 + d] + s;
    }
}

// ---------------- small helper kernels ----------------
__global__ void prep_w_kernel(const float* W, __bf16* out, int Kreal, int Kpad) {
    int idx = blockIdx.x * 256 + threadIdx.x;
    if (idx >= 128 * Kpad) return;
    int n = idx / Kpad; int k = idx - n * Kpad;
    out[idx] = (k < Kreal) ? (__bf16)W[(size_t)k * 128 + n] : (__bf16)0.f;
}

__global__ void node_feat_kernel(const float* pos, float* vel) {
    int n = blockIdx.x * 256 + threadIdx.x;
    if (n >= NNODE) return;
    const float* p = pos + (size_t)n * 18;
    float* v = vel + (size_t)n * 15;
#pragma unroll
    for (int s = 0; s < 5; ++s)
#pragma unroll
        for (int d = 0; d < 3; ++d) v[s * 3 + d] = p[(s + 1) * 3 + d] - p[s * 3 + d];
}

__global__ void edge_feat_kernel(const float* pos, float* ef, float* mask) {
    int e = blockIdx.x * 256 + threadIdx.x;
    if (e >= NEDGE) return;
    int b = e / 484; int rem = e - b * 484; int i = rem / 22; int j = rem - i * 22;
    const float* pi = pos + (size_t)(b * 22 + i) * 18 + 15;
    const float* pj = pos + (size_t)(b * 22 + j) * 18 + 15;
    float r0 = pi[0] - pj[0], r1 = pi[1] - pj[1], r2 = pi[2] - pj[2];
    float d = sqrtf(r0 * r0 + r1 * r1 + r2 * r2);
    float* o = ef + (size_t)e * 4;
    o[0] = r0; o[1] = r1; o[2] = r2; o[3] = d;
    mask[e] = (d <= 10.f) ? 1.f : 0.f;
}

__global__ void agg_kernel(const float* edge, float* agg) {
    int idx = blockIdx.x * 256 + threadIdx.x;
    if (idx >= NNODE * LAT) return;
    int n = idx >> 7; int c = idx & 127;
    int b = n / 22; int j = n - b * 22;
    const float* base = edge + ((size_t)b * 484 + j) * LAT + c;   // e = b*484 + i*22 + j
    float s = 0.f;
#pragma unroll
    for (int i = 0; i < 22; ++i) s += base[(size_t)i * 22 * LAT];
    agg[idx] = s;
}

// ---------------- host launcher ----------------
static constexpr size_t smem_bytes(int K1) {
    size_t R0 = ((size_t)32 * K1 * 2 > 16384) ? (size_t)32 * K1 * 2 : 16384;
    return R0 + (size_t)32 * 128 * 2 + 32 * 8 * 4 * 2 + 32 * 2 * 4;
}

extern "C" void kernel_launch(void* const* d_in, const int* in_sizes, int n_in,
                              void* d_out, int out_size, void* d_ws, size_t ws_size,
                              hipStream_t stream) {
    (void)n_in; (void)out_size; (void)ws_size;
    // ---- input index maps: jax-pytree sorted-key order OR python insertion order ----
    int decW[3], decB[3], eeW[3], eeB[3], neW[3], neB[3];
    int eelng, eelnb, nelng, nelnb;
    int seW[10][3], seB[10][3], snW[10][3], snB[10][3];
    int selng[10], selnb[10], snlng[10], snlnb[10];
    int posIdx;
    const bool insertion = (in_sizes[0] == NNODE * 6 * 3);  // position_sequence first?
    int i = 0;
    if (insertion) {
        posIdx = i++;
        for (int l = 0; l < 3; ++l) { neW[l] = i++; neB[l] = i++; }
        nelng = i++; nelnb = i++;
        for (int l = 0; l < 3; ++l) { eeW[l] = i++; eeB[l] = i++; }
        eelng = i++; eelnb = i++;
        for (int s = 0; s < 10; ++s) {
            for (int l = 0; l < 3; ++l) { seW[s][l] = i++; seB[s][l] = i++; }
            selng[s] = i++; selnb[s] = i++;
            for (int l = 0; l < 3; ++l) { snW[s][l] = i++; snB[s][l] = i++; }
            snlng[s] = i++; snlnb[s] = i++;
        }
        for (int l = 0; l < 3; ++l) { decW[l] = i++; decB[l] = i++; }
    } else {  // sorted keys: dec_mlp, edge_enc_ln, edge_enc_mlp, node_enc_ln, node_enc_mlp, steps
        for (int l = 0; l < 3; ++l) { decW[l] = i++; decB[l] = i++; }
        eelng = i++; eelnb = i++;
        for (int l = 0; l < 3; ++l) { eeW[l] = i++; eeB[l] = i++; }
        nelng = i++; nelnb = i++;
        for (int l = 0; l < 3; ++l) { neW[l] = i++; neB[l] = i++; }
        for (int s = 0; s < 10; ++s) {  // step keys sorted: edge_ln, edge_mlp, node_ln, node_mlp
            selng[s] = i++; selnb[s] = i++;
            for (int l = 0; l < 3; ++l) { seW[s][l] = i++; seB[s][l] = i++; }
            snlng[s] = i++; snlnb[s] = i++;
            for (int l = 0; l < 3; ++l) { snW[s][l] = i++; snB[s][l] = i++; }
        }
        posIdx = i++;
    }
    auto F = [&](int idx) -> const float* { return (const float*)d_in[idx]; };

    // ---- workspace carve (~76 MB total) ----
    char* ws = (char*)d_ws; size_t off = 0;
    auto alloc = [&](size_t bytes) -> void* {
        void* p = ws + off; off = (off + bytes + 255) & ~(size_t)255; return p;
    };
    float* vel   = (float*)alloc((size_t)NNODE * 15 * 4);
    float* efeat = (float*)alloc((size_t)NEDGE * 4 * 4);
    float* mask  = (float*)alloc((size_t)NEDGE * 4);
    float* node  = (float*)alloc((size_t)NNODE * LAT * 4);
    float* edge  = (float*)alloc((size_t)NEDGE * LAT * 4);
    float* agg   = (float*)alloc((size_t)NNODE * LAT * 4);

    auto prep = [&](int widx, int Kreal, int Kpad) -> __bf16* {
        __bf16* p = (__bf16*)alloc((size_t)128 * Kpad * 2);
        int total = 128 * Kpad;
        prep_w_kernel<<<(total + 255) / 256, 256, 0, stream>>>(F(widx), p, Kreal, Kpad);
        return p;
    };

    __bf16* ne_w[3] = { prep(neW[0], 15, 32),  prep(neW[1], 128, 128), prep(neW[2], 128, 128) };
    __bf16* ee_w[3] = { prep(eeW[0], 4, 32),   prep(eeW[1], 128, 128), prep(eeW[2], 128, 128) };
    __bf16* dc_w1 = prep(decW[0], 128, 128);
    __bf16* dc_w2 = prep(decW[1], 128, 128);
    __bf16 *se_w[10][3], *sn_w[10][3];
    for (int s = 0; s < 10; ++s) {
        se_w[s][0] = prep(seW[s][0], 384, 384);
        se_w[s][1] = prep(seW[s][1], 128, 128);
        se_w[s][2] = prep(seW[s][2], 128, 128);
        sn_w[s][0] = prep(snW[s][0], 256, 256);
        sn_w[s][1] = prep(snW[s][1], 128, 128);
        sn_w[s][2] = prep(snW[s][2], 128, 128);
    }

    // ---- features ----
    node_feat_kernel<<<(NNODE + 255) / 256, 256, 0, stream>>>(F(posIdx), vel);
    edge_feat_kernel<<<(NEDGE + 255) / 256, 256, 0, stream>>>(F(posIdx), efeat, mask);

    // ---- encoders ----
    mlp3_kernel<MODE_GENERIC, 32><<<NNODE / 32, 256, smem_bytes(32), stream>>>(
        vel, nullptr, 15, ne_w[0], F(neB[0]), ne_w[1], F(neB[1]), ne_w[2], F(neB[2]),
        F(nelng), F(nelnb), nullptr, nullptr, node);
    mlp3_kernel<MODE_GENERIC, 32><<<NEDGE / 32, 256, smem_bytes(32), stream>>>(
        efeat, nullptr, 4, ee_w[0], F(eeB[0]), ee_w[1], F(eeB[1]), ee_w[2], F(eeB[2]),
        F(eelng), F(eelnb), mask, nullptr, edge);

    // ---- message-passing steps ----
    for (int s = 0; s < NSTEPS; ++s) {
        mlp3_kernel<MODE_EDGE, 384><<<NEDGE / 32, 256, smem_bytes(384), stream>>>(
            edge, node, 384, se_w[s][0], F(seB[s][0]), se_w[s][1], F(seB[s][1]),
            se_w[s][2], F(seB[s][2]), F(selng[s]), F(selnb[s]), mask, edge, edge);
        agg_kernel<<<(NNODE * LAT + 255) / 256, 256, 0, stream>>>(edge, agg);
        mlp3_kernel<MODE_NODE, 256><<<NNODE / 32, 256, smem_bytes(256), stream>>>(
            node, agg, 256, sn_w[s][0], F(snB[s][0]), sn_w[s][1], F(snB[s][1]),
            sn_w[s][2], F(snB[s][2]), F(snlng[s]), F(snlnb[s]), nullptr, node, node);
    }

    // ---- decoder + position integrate ----
    decoder_kernel<<<NNODE / 32, 256, 0, stream>>>(
        node, dc_w1, F(decB[0]), dc_w2, F(decB[1]), F(decW[2]), F(decB[2]),
        F(posIdx), (float*)d_out);
}
